// MultiHeadDotProductAttention_62156766708103
// MI455X (gfx1250) — matrix operand order repaired and checked by reading, not compiled, not run on hardware
//
#include <hip/hip_runtime.h>

// ---------------------------------------------------------------------------
// MultiHeadDotProductAttention for MI455X (gfx1250, wave32, WMMA).
// Pipeline: 3x bf16-WMMA projection GEMMs -> flash-attention (online softmax,
// bf16 WMMA, f32 accum) -> bf16-WMMA output projection (f32 result).
// LDS double-buffering + CDNA5 async global->LDS path (ASYNCcnt +
// s_wait_asynccnt) so HBM tile fetches overlap WMMA compute.
// ---------------------------------------------------------------------------

typedef __attribute__((ext_vector_type(16))) __bf16 v16bf;
typedef __attribute__((ext_vector_type(8)))  float  v8f;

// Async-to-LDS builtin operand types: pointer-to-int4 in AS1 (global) / AS3
// (LDS); the pointee width encodes the b128 transfer size.
typedef __attribute__((__vector_size__(4 * sizeof(int)))) int i32x4;
typedef __attribute__((address_space(1))) i32x4* global_i32x4_p;
typedef __attribute__((address_space(3))) i32x4* lds_i32x4_p;

#define DEVINL __device__ __forceinline__

#if defined(__has_builtin)
#if __has_builtin(__builtin_amdgcn_global_load_async_to_lds_b128) && \
    __has_builtin(__builtin_amdgcn_s_wait_asynccnt)
#define HAVE_ASYNC_LDS 1
#endif
#endif
#ifndef HAVE_ASYNC_LDS
#define HAVE_ASYNC_LDS 0
#endif

constexpr int BATCH = 4;
constexpr int SEQ   = 2048;
constexpr int EMB   = 1024;
constexpr int NH    = 16;
constexpr int HD    = 64;
constexpr int HDIM  = NH * HD;      // 1024

// GEMM tiling: 128x128 block, K-slab 64, 8 waves (4M x 2N), wave = 32x64.
constexpr int TBM = 128, TBN = 128, TBK = 64;
constexpr int LDA_S = TBK + 8;      // 72 bf16 -> 144B row stride (16B aligned)
constexpr int LDB_S = TBN + 8;      // 136 bf16 -> 272B row stride (16B aligned)

enum { MODE_OUT_F32 = 0, MODE_Q = 1, MODE_KT = 2, MODE_V = 3 };

// ---- WMMA wrapper ----------------------------------------------------------
DEVINL v8f wmma_bf16(v16bf a, v16bf b, v8f c) {
  // D = A(16x32 bf16) * B(32x16 bf16) + C(16x16 f32)
  return __builtin_amdgcn_wmma_f32_16x16x32_bf16(
      /*neg_a=*/false, a, /*neg_b=*/false, b,
      /*c_mod=*/(short)0, c, /*reuse_a=*/false, /*reuse_b=*/false);
}

// ---- wait for this wave's outstanding async-to-LDS transfers ---------------
DEVINL void async_fence_all() {
#if HAVE_ASYNC_LDS
  __builtin_amdgcn_s_wait_asynccnt(0);
#endif
}

// ---- global(f32) -> LDS(bf16) converting tile copy (256 threads) -----------
DEVINL void tile_g2s_f32(__bf16* dst, int dld, const float* src, int sld,
                         int rows, int cols, int tid) {
  const int cpr = cols >> 2;
  const int chunks = rows * cpr;
  for (int c = tid; c < chunks; c += 256) {
    int r = c / cpr, cc = (c % cpr) << 2;
    float4 f = *(const float4*)(src + (size_t)r * sld + cc);
    union { __bf16 h[4]; uint2 u; } pk;
    pk.h[0] = (__bf16)f.x; pk.h[1] = (__bf16)f.y;
    pk.h[2] = (__bf16)f.z; pk.h[3] = (__bf16)f.w;
    *(uint2*)(dst + (size_t)r * dld + cc) = pk.u;
  }
}

// ---- global(bf16) -> LDS(bf16) tile copy: async DMA path if available ------
DEVINL void tile_g2s_bf16(__bf16* dst, int dld, const __bf16* src, int sld,
                          int rows, int cols, int tid) {
  const int cpr = cols >> 3;
  const int chunks = rows * cpr;
  for (int c = tid; c < chunks; c += 256) {
    int r = c / cpr, cc = (c % cpr) << 3;
#if HAVE_ASYNC_LDS
    // GLOBAL_LOAD_ASYNC_TO_LDS_B128: 16B per lane, LDS written by the async
    // engine, tracked with ASYNCcnt (no VGPR staging, no LOADcnt stall).
    __builtin_amdgcn_global_load_async_to_lds_b128(
        (global_i32x4_p)(src + (size_t)r * sld + cc),
        (lds_i32x4_p)(dst + (size_t)r * dld + cc),
        0, 0);
#else
    *(uint4*)(dst + (size_t)r * dld + cc) =
        *(const uint4*)(src + (size_t)r * sld + cc);
#endif
  }
}

// ---- fragment loaders (ISA 7.12.2 layouts, wave32) -------------------------
// A 16x32 bf16: lane<16 holds row M=lane, K 0-7 & 16-23; lane>=16 holds row
// M=lane-16, K 8-15 & 24-31. Two 16B reads per lane.
DEVINL v16bf load_a_frag(const __bf16* base, int ldr, int mrow, int kbase,
                         int lane) {
  const int lh = lane >> 4, r = lane & 15;
  const __bf16* p = base + (size_t)(mrow + r) * ldr + kbase + lh * 8;
  union { uint4 u[2]; v16bf v; } f;
  f.u[0] = *(const uint4*)p;
  f.u[1] = *(const uint4*)(p + 16);
  return f.v;
}

// B 32x16 bf16: lane holds column N=lane&15; lane<16 holds K 0-7 & 16-23,
// lane>=16 holds K 8-15 & 24-31 (matches A). Strided scalar reads per lane.
DEVINL v16bf load_b_frag(const __bf16* base, int ldr, int kbase, int ncol,
                         int lane) {
  const int lh = lane >> 4;
  const __bf16* p = base + (size_t)(kbase + lh * 8) * ldr + ncol + (lane & 15);
  v16bf v;
#pragma unroll
  for (int i = 0; i < 8; ++i) {
    v[i] = p[(size_t)i * ldr];
    v[8 + i] = p[(size_t)(16 + i) * ldr];
  }
  return v;
}

// ---------------------------------------------------------------------------
// GEMM: C(MxN) = A(MxK) * B(KxN), A f32-or-bf16 row-major, B f32 row-major.
// Double-buffered LDS slabs; one barrier per K-slab. MODE selects output:
//   MODE_OUT_F32 : f32 row-major (final projection -> d_out)
//   MODE_Q       : bf16 (b,h,q,d), scaled by 1/sqrt(64)
//   MODE_KT      : bf16 (b,h,d,kv)   (transposed K for attention B-fragments)
//   MODE_V       : bf16 (b,h,kv,d)
// ---------------------------------------------------------------------------
template<int MODE, typename AT>
__global__ __launch_bounds__(256) void proj_gemm_kernel(
    const AT* __restrict__ A, const float* __restrict__ Bw,
    float* __restrict__ outF, __bf16* __restrict__ outB,
    int M, int N, int K) {
  __shared__ alignas(16) __bf16 As[2][TBM * LDA_S];
  __shared__ alignas(16) __bf16 Bs[2][TBK * LDB_S];

  const int tid   = threadIdx.x;
  const int lane  = tid & 31;
  const int wv    = tid >> 5;       // 0..7
  const int waveM = wv >> 1;        // 0..3 -> 32-row slice
  const int waveN = wv & 1;         // 0..1 -> 64-col slice
  const int blockN = blockIdx.x * TBN;
  const int blockM = blockIdx.y * TBM;

  auto stage = [&](int buf, int kb) {
    if constexpr (sizeof(AT) == 2)
      tile_g2s_bf16(As[buf], LDA_S,
                    (const __bf16*)A + (size_t)blockM * K + kb, K,
                    TBM, TBK, tid);
    else
      tile_g2s_f32(As[buf], LDA_S,
                   (const float*)A + (size_t)blockM * K + kb, K,
                   TBM, TBK, tid);
    tile_g2s_f32(Bs[buf], LDB_S, Bw + (size_t)kb * N + blockN, N,
                 TBK, TBN, tid);
  };

  const v8f vzero = {};
  v8f acc[2][4];
#pragma unroll
  for (int mt = 0; mt < 2; ++mt)
#pragma unroll
    for (int nt = 0; nt < 4; ++nt) acc[mt][nt] = vzero;

  stage(0, 0);
  async_fence_all();
  __syncthreads();

  const int nslab = K / TBK;
  for (int si = 0; si < nslab; ++si) {
    const int cur = si & 1;
    if (si + 1 < nslab) stage(cur ^ 1, (si + 1) * TBK);  // overlaps compute

    const __bf16* Asb = As[cur];
    const __bf16* Bsb = Bs[cur];
#pragma unroll
    for (int kk = 0; kk < TBK; kk += 32) {
      v16bf bfr[4];
#pragma unroll
      for (int nt = 0; nt < 4; ++nt)
        bfr[nt] = load_b_frag(Bsb, LDB_S, kk, waveN * 64 + nt * 16, lane);
#pragma unroll
      for (int mt = 0; mt < 2; ++mt) {
        v16bf af = load_a_frag(Asb, LDA_S, waveM * 32 + mt * 16, kk, lane);
#pragma unroll
        for (int nt = 0; nt < 4; ++nt)
          acc[mt][nt] = wmma_bf16(af, bfr[nt], acc[mt][nt]);
      }
    }
    async_fence_all();
    __syncthreads();
  }

  // C/D layout: VGPR r -> row (r or r+8), N = lane&15.
  const int lh = lane >> 4, col = lane & 15;
#pragma unroll
  for (int mt = 0; mt < 2; ++mt)
#pragma unroll
    for (int nt = 0; nt < 4; ++nt)
#pragma unroll
      for (int r = 0; r < 8; ++r) {
        const int Mg = blockM + waveM * 32 + mt * 16 + lh * 8 + r;
        const int Ng = blockN + waveN * 64 + nt * 16 + col;
        const float v = acc[mt][nt][r];
        if constexpr (MODE == MODE_OUT_F32) {
          outF[(size_t)Mg * N + Ng] = v;
        } else {
          const int bb = Mg >> 11, pos = Mg & (SEQ - 1);   // M = b*SEQ + pos
          const int hh = Ng >> 6,  dd  = Ng & (HD - 1);    // N = h*HD + d
          if constexpr (MODE == MODE_Q) {
            outB[((size_t)(bb * NH + hh) * SEQ + pos) * HD + dd] =
                (__bf16)(v * 0.125f);                      // 1/sqrt(64)
          } else if constexpr (MODE == MODE_KT) {
            outB[((size_t)(bb * NH + hh) * HD + dd) * SEQ + pos] = (__bf16)v;
          } else {
            outB[((size_t)(bb * NH + hh) * SEQ + pos) * HD + dd] = (__bf16)v;
          }
        }
      }
}

// ---------------------------------------------------------------------------
// Flash attention: one WG per (b, h, 128 q rows); 8 waves x 16 q rows each.
// K/V chunks double-buffered with async-to-LDS loads; online softmax state
// lives per-lane for rows (r + 8*halfwave) and is reduced across the 16-lane
// N-group with shfl_xor (wave32-native).
// ---------------------------------------------------------------------------
__global__ __launch_bounds__(256) void flash_attn_kernel(
    const __bf16* __restrict__ Qb, const __bf16* __restrict__ Ktb,
    const __bf16* __restrict__ Vb, __bf16* __restrict__ Ob) {
  __shared__ alignas(16) __bf16 Qs[128 * 72];      // 128 q x 64 d
  __shared__ alignas(16) __bf16 Ks[2][64 * 72];    // 64 d x 64 kv (K^T slab)
  __shared__ alignas(16) __bf16 Vs[2][64 * 72];    // 64 kv x 64 d
  __shared__ alignas(16) __bf16 Ps[8][16 * 72];    // per-wave P staging 16x64

  const int tid = threadIdx.x, lane = tid & 31, wv = tid >> 5;
  const int lh = lane >> 4, col = lane & 15;
  const int qbase = blockIdx.x * 128;
  const int hh = blockIdx.y, bb = blockIdx.z;

  const size_t bh  = (size_t)(bb * NH + hh);
  const size_t qv0 = bh * SEQ;        // token-row base for Q / V
  const size_t kt0 = bh * HD * SEQ;   // row base for K^T (d-major)

  auto stage_kv = [&](int buf, int ckv) {
    tile_g2s_bf16(Ks[buf], 72, Ktb + kt0 + ckv,        SEQ, HD, 64, tid);
    tile_g2s_bf16(Vs[buf], 72, Vb + (qv0 + ckv) * HD,  HD,  64, HD, tid);
  };

  tile_g2s_bf16(Qs, 72, Qb + (qv0 + qbase) * HD, HD, 128, HD, tid);
  stage_kv(0, 0);
  async_fence_all();
  __syncthreads();

  const v8f vzero = {};
  float mstat[8], lstat[8];
  v8f oacc[4];
#pragma unroll
  for (int r = 0; r < 8; ++r) { mstat[r] = -INFINITY; lstat[r] = 0.f; }
#pragma unroll
  for (int dt = 0; dt < 4; ++dt) oacc[dt] = vzero;

  const int nchunk = SEQ / 64;   // 32
  for (int ci = 0; ci < nchunk; ++ci) {
    const int cur = ci & 1;
    if (ci + 1 < nchunk) stage_kv(cur ^ 1, (ci + 1) * 64);  // overlap compute

    // ---- S = Q * K^T : 16 q rows x 64 kv, K-dim = d = 64 ----
    v8f s[4];
#pragma unroll
    for (int nt = 0; nt < 4; ++nt) s[nt] = vzero;
#pragma unroll
    for (int kk = 0; kk < HD; kk += 32) {
      v16bf af = load_a_frag(Qs, 72, wv * 16, kk, lane);
#pragma unroll
      for (int nt = 0; nt < 4; ++nt) {
        v16bf bf = load_b_frag(Ks[cur], 72, kk, nt * 16, lane);
        s[nt] = wmma_bf16(af, bf, s[nt]);
      }
    }

    // ---- online softmax update ----
#pragma unroll
    for (int r = 0; r < 8; ++r) {
      float cm = fmaxf(fmaxf(s[0][r], s[1][r]), fmaxf(s[2][r], s[3][r]));
      cm = fmaxf(cm, __shfl_xor(cm, 1, 16));
      cm = fmaxf(cm, __shfl_xor(cm, 2, 16));
      cm = fmaxf(cm, __shfl_xor(cm, 4, 16));
      cm = fmaxf(cm, __shfl_xor(cm, 8, 16));
      const float mold = mstat[r];
      const float mnew = fmaxf(mold, cm);
      const float rescale = __expf(mold - mnew);   // first chunk: exp(-inf)=0
      mstat[r] = mnew;
      float rs = 0.f;
#pragma unroll
      for (int nt = 0; nt < 4; ++nt) {
        const float p = __expf(s[nt][r] - mnew);
        s[nt][r] = p;
        rs += p;
      }
      rs += __shfl_xor(rs, 1, 16);
      rs += __shfl_xor(rs, 2, 16);
      rs += __shfl_xor(rs, 4, 16);
      rs += __shfl_xor(rs, 8, 16);
      lstat[r] = lstat[r] * rescale + rs;
#pragma unroll
      for (int dt = 0; dt < 4; ++dt) oacc[dt][r] *= rescale;
    }

    // ---- P: C-layout -> row-major LDS (per-wave; LDS in-order per wave) ----
#pragma unroll
    for (int nt = 0; nt < 4; ++nt)
#pragma unroll
      for (int r = 0; r < 8; ++r)
        Ps[wv][(lh * 8 + r) * 72 + nt * 16 + col] = (__bf16)s[nt][r];

    // ---- O += P * V : K-dim = 64 kv ----
#pragma unroll
    for (int kk = 0; kk < 64; kk += 32) {
      v16bf af = load_a_frag(Ps[wv], 72, 0, kk, lane);
#pragma unroll
      for (int dt = 0; dt < 4; ++dt) {
        v16bf bf = load_b_frag(Vs[cur], 72, kk, dt * 16, lane);
        oacc[dt] = wmma_bf16(af, bf, oacc[dt]);
      }
    }
    async_fence_all();   // next chunk's tiles have landed in LDS
    __syncthreads();     // all waves done reading cur, loads visible WG-wide
  }

  // ---- normalize and store O as bf16 in (b, q, h, d) for final GEMM ----
#pragma unroll
  for (int r = 0; r < 8; ++r) {
    const float inv = 1.0f / lstat[r];
    const int q = qbase + wv * 16 + lh * 8 + r;
#pragma unroll
    for (int dt = 0; dt < 4; ++dt) {
      const float v = oacc[dt][r] * inv;
      Ob[(((size_t)bb * SEQ + q) * NH + hh) * HD + dt * 16 + col] = (__bf16)v;
    }
  }
}

// ---------------------------------------------------------------------------
extern "C" void kernel_launch(void* const* d_in, const int* in_sizes, int n_in,
                              void* d_out, int out_size, void* d_ws,
                              size_t ws_size, hipStream_t stream) {
  (void)in_sizes; (void)n_in; (void)out_size; (void)ws_size;
  const float* Xq  = (const float*)d_in[0];   // (4, 2048, 1024)
  const float* Xkv = (const float*)d_in[1];   // (4, 2048, 1024)
  const float* Wq  = (const float*)d_in[2];   // (1024, 16, 64) == (1024,1024)
  const float* Wk  = (const float*)d_in[3];
  const float* Wv  = (const float*)d_in[4];
  const float* Wo  = (const float*)d_in[5];   // (16, 64, 1024) == (1024,1024)
  float* out = (float*)d_out;                 // (4, 2048, 1024)

  // Workspace: 4 bf16 tensors of b*h*q*d = 8,388,608 elems each (64 MB total).
  const size_t nBHQD = (size_t)BATCH * NH * SEQ * HD;
  __bf16* Qb  = (__bf16*)d_ws;                // (b,h,q,d), pre-scaled
  __bf16* Ktb = Qb + nBHQD;                   // (b,h,d,kv)
  __bf16* Vb  = Ktb + nBHQD;                  // (b,h,kv,d)
  __bf16* Ob  = Vb + nBHQD;                   // (b,q,h,d)

  const int M = BATCH * SEQ;                  // 8192
  const dim3 gemmGrid(HDIM / TBN, M / TBM);   // (8, 64)

  proj_gemm_kernel<MODE_Q,  float><<<gemmGrid, 256, 0, stream>>>(
      Xq,  Wq, nullptr, Qb,  M, HDIM, EMB);
  proj_gemm_kernel<MODE_KT, float><<<gemmGrid, 256, 0, stream>>>(
      Xkv, Wk, nullptr, Ktb, M, HDIM, EMB);
  proj_gemm_kernel<MODE_V,  float><<<gemmGrid, 256, 0, stream>>>(
      Xkv, Wv, nullptr, Vb,  M, HDIM, EMB);

  flash_attn_kernel<<<dim3(SEQ / 128, NH, BATCH), 256, 0, stream>>>(
      Qb, Ktb, Vb, Ob);

  proj_gemm_kernel<MODE_OUT_F32, __bf16><<<dim3(EMB / TBN, M / TBM), 256, 0,
                                           stream>>>(
      Ob, Wo, out, nullptr, M, EMB, HDIM);
}
